// Model_60567628808244
// MI455X (gfx1250) — compile-verified, matrix operand order checked by reading, not run on hardware
//
#include <hip/hip_runtime.h>

// ---------------------------------------------------------------------------
// GCN 4-branch model on MI455X (gfx1250, wave32).
// Phase 1: one-shot fp32 -> bf16 conversion of all GEMM operands (v_perm pack).
// Phase 2: bf16 WMMA GEMM with double-buffered LDS, async global->LDS DMA for
//          the A tile (ASYNCcnt path), VGPR-transposed B staging, XOR-swizzled
//          LDS to keep ds_load_b128 conflict-free in exactly 64 KB.
// ---------------------------------------------------------------------------

typedef __attribute__((ext_vector_type(16))) __bf16 v16bf;
typedef __attribute__((ext_vector_type(8)))  float  v8f;
typedef __attribute__((ext_vector_type(4)))  unsigned int vu4;

#define BM 128
#define BN 128
#define BK 64

constexpr int ABYTES = BM * 128;           // A tile: 128 rows x 64 bf16 = 16 KB
constexpr int BUFSZ  = ABYTES + BN * 128;  // + B tile (transposed) 16 KB = 32 KB

union Frag {
    vu4 u[2];     // two 16-byte LDS loads
    v16bf bf;     // operand for v_wmma_f32_16x16x32_bf16
};

__device__ __forceinline__ unsigned bfround(float f) {
    // fp32 -> bf16 round-to-nearest-even, result in bits [31:16]
    unsigned u = __float_as_uint(f);
    return u + 0x7FFFu + ((u >> 16) & 1u);
}
__device__ __forceinline__ unsigned pack2(float lo, float hi) {
    // v_perm_b32: take high halves of both rounded values -> packed 2x bf16
    return __builtin_amdgcn_perm(bfround(hi), bfround(lo), 0x07060302u);
}
__device__ __forceinline__ unsigned swz8(int r) {
    // 16B-chunk XOR swizzle, conflict-free across 16 consecutive rows
    return (unsigned)((r ^ (r >> 3)) & 7);
}

// ---------------------------------------------------------------------------
// fp32 -> bf16 bulk conversion (8 elements / thread, b128 in / b128 out)
// ---------------------------------------------------------------------------
__global__ __launch_bounds__(256)
void cvt_f32_bf16(const float* __restrict__ src, unsigned short* __restrict__ dst, long n)
{
    long i = ((long)blockIdx.x * 256 + threadIdx.x) * 8;
    if (i >= n) return;
    float4 f0 = *(const float4*)(src + i);
    float4 f1 = *(const float4*)(src + i + 4);
    uint4 o;
    o.x = pack2(f0.x, f0.y);
    o.y = pack2(f0.z, f0.w);
    o.z = pack2(f1.x, f1.y);
    o.w = pack2(f1.z, f1.w);
    *(uint4*)(dst + i) = o;
}

// ---------------------------------------------------------------------------
// C[M,N] = A[M,K] * B[K,N] (+bias) (+relu); A,B bf16 row-major.
// out_bf16=1 -> C is bf16 (intermediates), else fp32 (final outputs).
// ---------------------------------------------------------------------------
__global__ __launch_bounds__(256)
void gemm_bf16_wmma(const unsigned short* __restrict__ A,
                    const unsigned short* __restrict__ B,
                    const float* __restrict__ bias,
                    void* __restrict__ Cout,
                    int M, int N, int K, int relu, int out_bf16)
{
    __shared__ __align__(16) unsigned char smem[2 * BUFSZ];  // 64 KB, double buffered

    const int tid  = threadIdx.x;
    const int lane = tid & 31;
    const int wid  = tid >> 5;
    const int wm   = wid & 3;      // wave row    *32
    const int wn   = wid >> 2;     // wave column *64
    const long blockRow = (long)blockIdx.y * BM;
    const long blockCol = (long)blockIdx.x * BN;

    // A staging: thread -> (row, half-row of 4 x 16B chunks)
    const int arow  = tid >> 1;
    const int achnk = (tid & 1) * 4;
    const unsigned aswz = swz8(arow);
    const unsigned short* aG = A + (blockRow + arow) * (long)K + (tid & 1) * 32;

    // B staging: thread -> (4 consecutive k rows, 8 columns), stored transposed
    const int bk0 = (tid & 15) * 4;
    const int bn8 = (tid >> 4) * 8;
    const unsigned short* bG = B + (long)bk0 * N + blockCol + bn8;

    const int lr    = lane & 15;
    const int khalf = lane >> 4;

    v8f acc[2][4];
#pragma unroll
    for (int mi = 0; mi < 2; ++mi)
#pragma unroll
        for (int ni = 0; ni < 4; ++ni)
            acc[mi][ni] = {};

    uint4 bR[4];

    // ---- async DMA of A tile straight into LDS (ASYNCcnt) ----
    auto issueA = [&](int kt, int buf) {
        unsigned lbase = (unsigned)(uintptr_t)(smem + buf * BUFSZ + arow * 128);
        unsigned long long g = (unsigned long long)(uintptr_t)(aG + kt);
#pragma unroll
        for (int c = 0; c < 4; ++c) {
            unsigned lofs = lbase + (((unsigned)(achnk + c) ^ aswz) << 4);
            unsigned long long ga = g + (unsigned)(c * 16);
            asm volatile("global_load_async_to_lds_b128 %0, %1, off"
                         :: "v"(lofs), "v"(ga) : "memory");
        }
    };
    auto loadB = [&](int kt) {
        const unsigned short* g = bG + (long)kt * N;
#pragma unroll
        for (int j = 0; j < 4; ++j)
            bR[j] = *(const uint4*)(g + (long)j * N);
    };
    auto storeB = [&](int buf) {
        unsigned char* base = smem + buf * BUFSZ + ABYTES;
        const unsigned a0[4] = {bR[0].x, bR[0].y, bR[0].z, bR[0].w};
        const unsigned a1[4] = {bR[1].x, bR[1].y, bR[1].z, bR[1].w};
        const unsigned a2[4] = {bR[2].x, bR[2].y, bR[2].z, bR[2].w};
        const unsigned a3[4] = {bR[3].x, bR[3].y, bR[3].z, bR[3].w};
        const int d2  = (bk0 * 2) >> 4;   // 16B chunk within row
        const int sub = (bk0 * 2) & 15;   // 0 or 8
#pragma unroll
        for (int d = 0; d < 4; ++d) {
            // interleave rows k..k+3 into (k,k+1) and (k+2,k+3) bf16 pairs
            unsigned p00 = __builtin_amdgcn_perm(a1[d], a0[d], 0x05040100u);
            unsigned p01 = __builtin_amdgcn_perm(a1[d], a0[d], 0x07060302u);
            unsigned p10 = __builtin_amdgcn_perm(a3[d], a2[d], 0x05040100u);
            unsigned p11 = __builtin_amdgcn_perm(a3[d], a2[d], 0x07060302u);
            const int n0 = bn8 + 2 * d;
            uint2 q0; q0.x = p00; q0.y = p10;
            uint2 q1; q1.x = p01; q1.y = p11;
            *(uint2*)(base + n0 * 128 + (((unsigned)d2 ^ swz8(n0)) << 4) + sub) = q0;
            *(uint2*)(base + (n0 + 1) * 128 + (((unsigned)d2 ^ swz8(n0 + 1)) << 4) + sub) = q1;
        }
    };
    auto compute = [&](int buf) {
        const unsigned char* bufA = smem + buf * BUFSZ;
        const unsigned char* bufB = bufA + ABYTES;
#pragma unroll
        for (int s = 0; s < 2; ++s) {     // two 16x16x32 k-steps per BK=64 tile
            Frag fa[2], fb[4];
            const int d0 = s * 4 + khalf;
#pragma unroll
            for (int mi = 0; mi < 2; ++mi) {
                const int r = wm * 32 + mi * 16 + lr;
                const unsigned sw = swz8(r);
                const unsigned char* pa = bufA + r * 128;
                fa[mi].u[0] = *(const vu4*)(pa + (((unsigned)d0 ^ sw) << 4));
                fa[mi].u[1] = *(const vu4*)(pa + (((unsigned)(d0 + 2) ^ sw) << 4));
            }
#pragma unroll
            for (int ni = 0; ni < 4; ++ni) {
                const int r = wn * 64 + ni * 16 + lr;
                const unsigned sw = swz8(r);
                const unsigned char* pb = bufB + r * 128;
                fb[ni].u[0] = *(const vu4*)(pb + (((unsigned)d0 ^ sw) << 4));
                fb[ni].u[1] = *(const vu4*)(pb + (((unsigned)(d0 + 2) ^ sw) << 4));
            }
#pragma unroll
            for (int mi = 0; mi < 2; ++mi)
#pragma unroll
                for (int ni = 0; ni < 4; ++ni)
                    acc[mi][ni] = __builtin_amdgcn_wmma_f32_16x16x32_bf16(
                        false, fa[mi].bf, false, fb[ni].bf,
                        (short)0, acc[mi][ni], false, false);
        }
    };

    // ---- prologue: stage tile 0 ----
    issueA(0, 0);
    loadB(0);
    storeB(0);
    asm volatile("s_wait_asynccnt 0x0" ::: "memory");
    __syncthreads();

    // ---- main pipeline: one barrier per BK tile, DMA overlaps WMMA ----
    int cur = 0;
    for (int kt = 0; kt < K; kt += BK, cur ^= 1) {
        const bool hasNext = (kt + BK) < K;
        if (hasNext) {
            issueA(kt + BK, cur ^ 1);
            loadB(kt + BK);
            if (kt + 2 * BK < K) {          // global_prefetch_b8 into GL2
                __builtin_prefetch(aG + kt + 2 * BK, 0, 1);
                __builtin_prefetch(bG + (long)(kt + 2 * BK) * N, 0, 1);
            }
        }
        compute(cur);
        if (hasNext) storeB(cur ^ 1);
        asm volatile("s_wait_asynccnt 0x0" ::: "memory");
        __syncthreads();
    }

    // ---- epilogue: bias + optional relu, documented C/D layout ----
    if (out_bf16) {
        unsigned short* C = (unsigned short*)Cout;
#pragma unroll
        for (int ni = 0; ni < 4; ++ni) {
            const long col = blockCol + wn * 64 + ni * 16 + lr;
            const float bv = bias ? bias[col] : 0.0f;
#pragma unroll
            for (int mi = 0; mi < 2; ++mi) {
                const long row0 = blockRow + wm * 32 + mi * 16 + khalf * 8;
#pragma unroll
                for (int i = 0; i < 8; ++i) {
                    float v = acc[mi][ni][i] + bv;
                    if (relu) v = fmaxf(v, 0.0f);
                    C[(row0 + i) * (long)N + col] = (unsigned short)(bfround(v) >> 16);
                }
            }
        }
    } else {
        float* C = (float*)Cout;
#pragma unroll
        for (int ni = 0; ni < 4; ++ni) {
            const long col = blockCol + wn * 64 + ni * 16 + lr;
            const float bv = bias ? bias[col] : 0.0f;
#pragma unroll
            for (int mi = 0; mi < 2; ++mi) {
                const long row0 = blockRow + wm * 32 + mi * 16 + khalf * 8;
#pragma unroll
                for (int i = 0; i < 8; ++i) {
                    float v = acc[mi][ni][i] + bv;
                    if (relu) v = fmaxf(v, 0.0f);
                    C[(row0 + i) * (long)N + col] = v;
                }
            }
        }
    }
}

// ---------------------------------------------------------------------------
// Host-side orchestration
// ---------------------------------------------------------------------------
static inline void launch_cvt(const float* src, unsigned short* dst, long n,
                              hipStream_t stream)
{
    long blocks = (n / 8 + 255) / 256;
    hipLaunchKernelGGL(cvt_f32_bf16, dim3((unsigned)blocks), dim3(256), 0, stream,
                       src, dst, n);
}

static inline void launch_gemm(const unsigned short* A, const unsigned short* B,
                               const float* bias, void* C,
                               int M, int N, int K, int relu, int out_bf16,
                               hipStream_t stream)
{
    dim3 grid(N / BN, M / BM);
    hipLaunchKernelGGL(gemm_bf16_wmma, grid, dim3(256), 0, stream,
                       A, B, bias, C, M, N, K, relu, out_bf16);
}

extern "C" void kernel_launch(void* const* d_in, const int* in_sizes, int n_in,
                              void* d_out, int out_size, void* d_ws, size_t ws_size,
                              hipStream_t stream)
{
    (void)in_sizes; (void)n_in; (void)out_size; (void)ws_size;

    constexpr long N_  = 8192;
    constexpr long FIN = 512;
    constexpr long HID = 256;
    constexpr long H2  = 512;

    const float* x1   = (const float*)d_in[0];
    const float* x2   = (const float*)d_in[1];
    const float* adj1 = (const float*)d_in[2];
    const float* adj2 = (const float*)d_in[3];
    const float* adjd = (const float*)d_in[4];
    const float* W0   = (const float*)d_in[5];
    const float* b0   = (const float*)d_in[6];
    const float* W1   = (const float*)d_in[7];
    const float* b1   = (const float*)d_in[8];
    float* out = (float*)d_out;

    // bf16 workspace layout
    unsigned short* p = (unsigned short*)d_ws;
    unsigned short* adj1b = p; p += N_ * N_;
    unsigned short* adj2b = p; p += N_ * N_;
    unsigned short* adjdb = p; p += N_ * N_;
    unsigned short* x1b   = p; p += N_ * FIN;
    unsigned short* x2b   = p; p += N_ * FIN;
    unsigned short* W0b   = p; p += FIN * H2;
    unsigned short* W1b   = p; p += H2 * HID;
    unsigned short* s0x1b = p; p += N_ * H2;
    unsigned short* s0x2b = p; p += N_ * H2;
    unsigned short* hb    = p; p += N_ * H2;
    unsigned short* s1b   = p; p += N_ * HID;

    // Phase 1: one-shot conversions (adj1 reused by 4 GEMMs, adj2/adjd by 2)
    launch_cvt(adj1, adj1b, N_ * N_, stream);
    launch_cvt(adj2, adj2b, N_ * N_, stream);
    launch_cvt(adjd, adjdb, N_ * N_, stream);
    launch_cvt(x1,   x1b,   N_ * FIN, stream);
    launch_cvt(x2,   x2b,   N_ * FIN, stream);
    launch_cvt(W0,   W0b,   FIN * H2, stream);
    launch_cvt(W1,   W1b,   H2 * HID, stream);

    // Phase 2: shared support0 GEMMs (computed once per feature matrix)
    launch_gemm(x1b, W0b, nullptr, s0x1b, (int)N_, (int)H2, (int)FIN, 0, 1, stream);
    launch_gemm(x2b, W0b, nullptr, s0x2b, (int)N_, (int)H2, (int)FIN, 0, 1, stream);

    for (int b = 0; b < 4; ++b) {
        const unsigned short* adjb = (b == 2) ? adj2b : (b == 3) ? adjdb : adj1b;
        const unsigned short* s0b  = (b == 1) ? s0x2b : s0x1b;
        // h = relu(adj @ s0 + b0)           (bf16 out)
        launch_gemm(adjb, s0b, b0, hb, (int)N_, (int)H2, (int)N_, 1, 1, stream);
        // s1 = h @ W1                        (bf16 out)
        launch_gemm(hb, W1b, nullptr, s1b, (int)N_, (int)HID, (int)H2, 0, 1, stream);
        // out_b = adj @ s1 + b1              (fp32 out)
        launch_gemm(adjb, s1b, b1, out + (size_t)b * N_ * HID,
                    (int)N_, (int)HID, (int)N_, 0, 0, stream);
    }
}
